// MultiHeadedAttentionLayer_87780541596006
// MI455X (gfx1250) — compile-verified, matrix operand order checked by reading, not run on hardware
//
#include <hip/hip_runtime.h>
#include <hip/hip_bf16.h>

typedef __attribute__((ext_vector_type(16))) _Float16 v16h;
typedef __attribute__((ext_vector_type(8)))  _Float16 v8h;
typedef __attribute__((ext_vector_type(8)))  float    v8f;

#define Hh 12
#define Dd 64
#define BLKSZ 64
#define Rr 3
#define Bb 2
#define Ss 4096
#define Wd 768
#define Mm 64
#define NEGV (-1e9f)
#define KVPAD 72   // padded LDS row stride (halves): 144B, 16B-aligned, spreads banks

// A-frag (16x32 f16, row-major source) per ISA 7.12.2:
// lane l holds row l%16; elements 0..7 = K {kb..kb+7}, 8..15 = K {kb+16..kb+23}, kb = 8*(l>=16)
__device__ inline v16h load_frag_rowmajor(const _Float16* base, int ld, int row, int k0, int lane) {
  int r  = lane & 15;
  int kb = (lane & 16) ? 8 : 0;
  const _Float16* p = base + (size_t)(row + r) * ld + k0 + kb;
  v8h lo = *(const v8h*)p;
  v8h hi = *(const v8h*)(p + 16);
  v16h o;
#pragma unroll
  for (int i = 0; i < 8; ++i) { o[i] = lo[i]; o[8 + i] = hi[i]; }
  return o;
}

// B-frag (32x16 f16) from a row-major [K][N] source (column fixed per lane, strided K reads)
__device__ inline v16h load_frag_colmajor(const _Float16* base, int ld, int k0, int col, int lane) {
  int n  = lane & 15;
  int kb = (lane & 16) ? 8 : 0;
  const _Float16* p = base + (size_t)(k0 + kb) * ld + col + n;
  v16h o;
#pragma unroll
  for (int i = 0; i < 8; ++i) {
    o[i]     = p[(size_t)i * ld];
    o[8 + i] = p[(size_t)(16 + i) * ld];
  }
  return o;
}

// Issue coalesced async copies of one 64x64 f16 K block and V block into LDS
// (ASYNCcnt-tracked; 16B per thread per matrix, 512 threads cover 8KB each).
__device__ inline void async_load_kv(const _Float16* kbase, const _Float16* vbase,
                                     unsigned ldsK, unsigned ldsV, int tid) {
  int r = tid >> 3;                  // 0..63 rows
  int c = tid & 7;                   // 8 x 16B chunks per row
  const _Float16* gk = kbase + (size_t)r * Dd + c * 8;
  const _Float16* gv = vbase + (size_t)r * Dd + c * 8;
  unsigned lk = ldsK + (unsigned)(r * KVPAD + c * 8) * 2;
  unsigned lv = ldsV + (unsigned)(r * KVPAD + c * 8) * 2;
  asm volatile("global_load_async_to_lds_b128 %0, %1, off" :: "v"(lk), "v"(gk) : "memory");
  asm volatile("global_load_async_to_lds_b128 %0, %1, off" :: "v"(lv), "v"(gv) : "memory");
}

// ---------------------------------------------------------------------------
// Kernel 0: fp32 -> f16 conversions; weights transposed so B-operand reads
// are contiguous along K.
// ---------------------------------------------------------------------------
__global__ void convert_k(const float* __restrict__ from, const float* __restrict__ to,
                          const float* __restrict__ Wq, const float* __restrict__ Wk,
                          const float* __restrict__ Wv,
                          _Float16* __restrict__ Aq, _Float16* __restrict__ Ak,
                          _Float16* __restrict__ Wqt, _Float16* __restrict__ Wkt,
                          _Float16* __restrict__ Wvt) {
  size_t i = (size_t)blockIdx.x * blockDim.x + threadIdx.x;
  const size_t AE = (size_t)Bb * Ss * Wd;
  if (i < AE) {
    Aq[i] = (_Float16)from[i];
    Ak[i] = (_Float16)to[i];
  }
  if (i < (size_t)Wd * Wd) {
    int n = (int)(i / Wd), k = (int)(i % Wd);
    Wqt[i] = (_Float16)Wq[(size_t)k * Wd + n];
    Wkt[i] = (_Float16)Wk[(size_t)k * Wd + n];
    Wvt[i] = (_Float16)Wv[(size_t)k * Wd + n];
  }
}

// ---------------------------------------------------------------------------
// Kernel 1: QKV projection GEMM via WMMA. blockIdx.y selects q/k/v.
// Each wave computes one 16x16 tile of C = A[8192x768] * W[768x768] + bias,
// scattering the result into [B,H,S,D] f16 layout.
// ---------------------------------------------------------------------------
__global__ __launch_bounds__(256) void qkv_gemm(
    const _Float16* __restrict__ Aq, const _Float16* __restrict__ Ak,
    const _Float16* __restrict__ Wqt, const _Float16* __restrict__ Wkt,
    const _Float16* __restrict__ Wvt,
    const float* __restrict__ bq, const float* __restrict__ bk, const float* __restrict__ bv,
    _Float16* __restrict__ qb, _Float16* __restrict__ kb, _Float16* __restrict__ vb) {
  int lane = threadIdx.x & 31;
  int wave = threadIdx.x >> 5;
  int gw = blockIdx.x * 8 + wave;       // 24576 tiles = 512 (rows) x 48 (cols)
  int tile_m = gw & 511;
  int tile_n = gw >> 9;
  int which = blockIdx.y;

  const _Float16* A    = (which == 0) ? Aq : Ak;
  const _Float16* Wt   = (which == 0) ? Wqt : (which == 1) ? Wkt : Wvt;
  const float*    bias = (which == 0) ? bq  : (which == 1) ? bk  : bv;
  _Float16*       outp = (which == 0) ? qb  : (which == 1) ? kb  : vb;

  v8f acc = {};
#pragma unroll 4
  for (int kk = 0; kk < Wd; kk += 32) {
    v16h a = load_frag_rowmajor(A,  Wd, tile_m * 16, kk, lane);
    v16h b = load_frag_rowmajor(Wt, Wd, tile_n * 16, kk, lane);  // Wt row = output col n
    acc = __builtin_amdgcn_wmma_f32_16x16x32_f16(false, a, false, b, (short)0, acc, false, false);
  }

  int col = tile_n * 16 + (lane & 15);
  int hi8 = (lane & 16) ? 8 : 0;
  float bia = bias[col];
  int hh = col >> 6, dd = col & 63;
#pragma unroll
  for (int r = 0; r < 8; ++r) {
    int bs = tile_m * 16 + r + hi8;
    int b_ = bs >> 12, s = bs & (Ss - 1);
    outp[(((size_t)b_ * Hh + hh) * Ss + s) * Dd + dd] = (_Float16)(acc[r] + bia);
  }
}

// ---------------------------------------------------------------------------
// Kernel 2: BigBird block-sparse attention, flash-style online softmax.
// One workgroup (512 threads = 16 waves) per (b, h, query block m).
// K/V blocks staged to LDS with double-buffered global_load_async_to_lds_b128.
// Wave w: qt = w/4 (query 16-row slab), xt = w%4 (key / D 16-col slab).
// ---------------------------------------------------------------------------
__global__ __launch_bounds__(512) void bigbird_attn(
    const _Float16* __restrict__ qb, const _Float16* __restrict__ kb_g,
    const _Float16* __restrict__ vb,
    const int* __restrict__ to_mask, const int* __restrict__ rand_attn,
    float* __restrict__ out) {
  __shared__ _Float16 smem_k[2][64][KVPAD]; // double-buffered K block
  __shared__ _Float16 smem_v[2][64][KVPAD]; // double-buffered V block
  __shared__ float    smem_s[64][64];       // scores (fp32)
  __shared__ _Float16 smem_p[64][64];       // softmax probs (f16) for PV WMMA
  __shared__ float    red[64][8];
  __shared__ float    rowmax[64];
  __shared__ float    rowsum[64];
  __shared__ float    corr[64];
  __shared__ int      kblist[64];
  __shared__ int      nkb;

  int tid  = threadIdx.x;
  int lane = tid & 31;
  int wave = tid >> 5;
  int qt = wave >> 2;
  int xt = wave & 3;

  int bid = blockIdx.x;
  int b   = bid / (Hh * Mm);
  int rem = bid % (Hh * Mm);
  int h   = rem / Mm;
  int m   = rem % Mm;

  if (tid == 0) {
    int n = 0;
    if (m == 0 || m == Mm - 1) {
      for (int i = 0; i < Mm; ++i) kblist[n++] = i;           // dense row
    } else if (m == 1) {
      kblist[0] = 0; kblist[1] = 1; kblist[2] = 2; kblist[3] = Mm - 1; n = 4;
      for (int r = 0; r < Rr; ++r)
        kblist[n++] = rand_attn[((size_t)h * (Mm - 2) + (m - 1)) * Rr + r];
    } else if (m == Mm - 2) {
      kblist[0] = 0; kblist[1] = Mm - 3; kblist[2] = Mm - 2; kblist[3] = Mm - 1; n = 4;
      for (int r = 0; r < Rr; ++r)
        kblist[n++] = rand_attn[((size_t)h * (Mm - 2) + (m - 1)) * Rr + r];
    } else {
      kblist[0] = m - 1; kblist[1] = m; kblist[2] = m + 1;
      kblist[3] = 0; kblist[4] = Mm - 1; n = 5;
      for (int r = 0; r < Rr; ++r)
        kblist[n++] = rand_attn[((size_t)h * (Mm - 2) + (m - 1)) * Rr + r];
    }
    nkb = n;
  }
  if (tid < 64) { rowmax[tid] = -1e30f; rowsum[tid] = 0.f; }

  const size_t bh_off = ((size_t)b * Hh + h) * Ss;
  const _Float16* qbase = qb + (bh_off + m * BLKSZ) * Dd;
  v16h a0 = load_frag_rowmajor(qbase, Dd, qt * 16, 0, lane);
  v16h a1 = load_frag_rowmajor(qbase, Dd, qt * 16, 32, lane);
  v8f acc = {};
  __syncthreads();
  int nblocks = nkb;

  unsigned ldsK0 = (unsigned)(size_t)&smem_k[0][0][0];
  unsigned ldsK1 = (unsigned)(size_t)&smem_k[1][0][0];
  unsigned ldsV0 = (unsigned)(size_t)&smem_v[0][0][0];
  unsigned ldsV1 = (unsigned)(size_t)&smem_v[1][0][0];

  // prefetch key block 0 into buffer 0
  {
    int kblk = kblist[0];
    async_load_kv(kb_g + (bh_off + kblk * BLKSZ) * Dd,
                  vb   + (bh_off + kblk * BLKSZ) * Dd, ldsK0, ldsV0, tid);
  }

  for (int kbi = 0; kbi < nblocks; ++kbi) {
    int cur = kbi & 1;
    int kblk = kblist[kbi];

    // prefetch next block into the alternate buffer, then wait for current
    if (kbi + 1 < nblocks) {
      int nblk = kblist[kbi + 1];
      async_load_kv(kb_g + (bh_off + nblk * BLKSZ) * Dd,
                    vb   + (bh_off + nblk * BLKSZ) * Dd,
                    cur ? ldsK0 : ldsK1, cur ? ldsV0 : ldsV1, tid);
      asm volatile("s_wait_asynccnt 0x2" ::: "memory");
    } else {
      asm volatile("s_wait_asynccnt 0x0" ::: "memory");
    }
    __syncthreads();   // current K/V visible to all waves

    const _Float16* kl = &smem_k[cur][0][0];
    const _Float16* vl = &smem_v[cur][0][0];

    // S = Q * K^T : B-frag column = key row -> contiguous LDS reads along D
    v16h b0 = load_frag_rowmajor(kl, KVPAD, xt * 16, 0, lane);
    v16h b1 = load_frag_rowmajor(kl, KVPAD, xt * 16, 32, lane);
    v8f s = {};
    s = __builtin_amdgcn_wmma_f32_16x16x32_f16(false, a0, false, b0, (short)0, s, false, false);
    s = __builtin_amdgcn_wmma_f32_16x16x32_f16(false, a1, false, b1, (short)0, s, false, false);

    int keycol = xt * 16 + (lane & 15);
    float maskadd = (1.0f - (float)to_mask[(size_t)b * Ss + kblk * BLKSZ + keycol]) * NEGV;
    int hi8 = (lane & 16) ? 8 : 0;
#pragma unroll
    for (int r = 0; r < 8; ++r)
      smem_s[qt * 16 + r + hi8][keycol] = s[r] * 0.125f + maskadd;
    __syncthreads();

    // ---- online softmax over this 64-key chunk (8 threads per row) ----
    int row = tid >> 3;
    int c8  = (tid & 7) * 8;
    float mx = -1e30f;
#pragma unroll
    for (int i = 0; i < 8; ++i) mx = fmaxf(mx, smem_s[row][c8 + i]);
    red[row][tid & 7] = mx;
    __syncthreads();
    if ((tid & 7) == 0) {
      float bm = red[row][0];
#pragma unroll
      for (int i = 1; i < 8; ++i) bm = fmaxf(bm, red[row][i]);
      float mo = rowmax[row];
      float mn = fmaxf(mo, bm);
      corr[row]   = __expf(mo - mn);
      rowmax[row] = mn;
    }
    __syncthreads();
    float mn = rowmax[row];
    float psum = 0.f;
#pragma unroll
    for (int i = 0; i < 8; ++i) {
      float p = __expf(smem_s[row][c8 + i] - mn);
      smem_p[row][c8 + i] = (_Float16)p;
      psum += p;
    }
    red[row][tid & 7] = psum;
    __syncthreads();
    if ((tid & 7) == 0) {
      float sm = 0.f;
#pragma unroll
      for (int i = 0; i < 8; ++i) sm += red[row][i];
      rowsum[row] = rowsum[row] * corr[row] + sm;
    }
    __syncthreads();

    // ---- rescale accumulator, then O += P * V via WMMA (V from LDS) ----
#pragma unroll
    for (int r = 0; r < 8; ++r) acc[r] *= corr[qt * 16 + r + hi8];
    const _Float16* pp = &smem_p[0][0];
    v16h pa0 = load_frag_rowmajor(pp, 64, qt * 16, 0, lane);
    v16h pa1 = load_frag_rowmajor(pp, 64, qt * 16, 32, lane);
    v16h vb0 = load_frag_colmajor(vl, KVPAD, 0,  xt * 16, lane);
    v16h vb1 = load_frag_colmajor(vl, KVPAD, 32, xt * 16, lane);
    acc = __builtin_amdgcn_wmma_f32_16x16x32_f16(false, pa0, false, vb0, (short)0, acc, false, false);
    acc = __builtin_amdgcn_wmma_f32_16x16x32_f16(false, pa1, false, vb1, (short)0, acc, false, false);
    __syncthreads();   // protect smem_p / corr / LDS buffers before next iteration
  }

  // normalize and store fp32 [B,S,H,D]
  int hi8 = (lane & 16) ? 8 : 0;
  int d = xt * 16 + (lane & 15);
#pragma unroll
  for (int r = 0; r < 8; ++r) {
    int row = qt * 16 + r + hi8;
    out[(((size_t)b * Ss + m * BLKSZ + row) * Hh + h) * Dd + d] = acc[r] / rowsum[row];
  }
}

// ---------------------------------------------------------------------------
extern "C" void kernel_launch(void* const* d_in, const int* in_sizes, int n_in,
                              void* d_out, int out_size, void* d_ws, size_t ws_size,
                              hipStream_t stream) {
  (void)in_sizes; (void)n_in; (void)out_size; (void)ws_size;
  const float* from = (const float*)d_in[0];
  const float* to   = (const float*)d_in[1];
  const float* Wq   = (const float*)d_in[2];
  const float* bq   = (const float*)d_in[3];
  const float* Wk   = (const float*)d_in[4];
  const float* bk   = (const float*)d_in[5];
  const float* Wv   = (const float*)d_in[6];
  const float* bv   = (const float*)d_in[7];
  const int* to_mask   = (const int*)d_in[8];
  const int* rand_attn = (const int*)d_in[9];
  float* out = (float*)d_out;

  const size_t A_ELEMS = (size_t)Bb * Ss * Wd;   // 6291456
  const size_t W_ELEMS = (size_t)Wd * Wd;        // 589824
  _Float16* ws  = (_Float16*)d_ws;
  _Float16* Aq  = ws;
  _Float16* Ak  = Aq  + A_ELEMS;
  _Float16* Wqt = Ak  + A_ELEMS;
  _Float16* Wkt = Wqt + W_ELEMS;
  _Float16* Wvt = Wkt + W_ELEMS;
  _Float16* qb  = Wvt + W_ELEMS;
  _Float16* kb  = qb  + A_ELEMS;
  _Float16* vb  = kb  + A_ELEMS;

  int cvt_blocks = (int)((A_ELEMS + 255) / 256);
  convert_k<<<cvt_blocks, 256, 0, stream>>>(from, to, Wq, Wk, Wv, Aq, Ak, Wqt, Wkt, Wvt);

  dim3 g1(3072, 3);  // 24576 tiles / 8 waves-per-block, x3 for q/k/v
  qkv_gemm<<<g1, 256, 0, stream>>>(Aq, Ak, Wqt, Wkt, Wvt, bq, bk, bv, qb, kb, vb);

  bigbird_attn<<<Bb * Hh * Mm, 512, 0, stream>>>(qb, kb, vb, to_mask, rand_attn, out);
}
